// PointConvTransFlow_9354438770926
// MI455X (gfx1250) — matrix operand order repaired and checked by reading, not compiled, not run on hardware
//
#include <hip/hip_runtime.h>
#include <hip/hip_bf16.h>

typedef __attribute__((ext_vector_type(16))) _Float16 v16h;
typedef __attribute__((ext_vector_type(8)))  _Float16 v8h;
typedef __attribute__((ext_vector_type(8)))  float    v8f;

#define KNN_K     16
#define CM        64
#define DIM_IN    131      // 2*D + 3
#define DIM_PAD   160      // padded to 5 chunks of 32
#define LEAKY_S   0.1f
#define BN_SCALE  0.9999950000374997f   // 1/sqrt(1+1e-5)
#define INV_SQRT3 0.5773502691896258f

// ---------------------------------------------------------------------------
// Kernel 1: brute force KNN (query p1 against ref), xyz layout [B,3,N]
// one thread per query point, ref tiled through LDS, sorted top-16 in regs
// ---------------------------------------------------------------------------
__global__ void __launch_bounds__(256)
knn_kernel(const float* __restrict__ qxyz, const float* __restrict__ rxyz,
           int* __restrict__ outidx, int N)
{
    const int flat = blockIdx.x * 256 + threadIdx.x;   // b*N + n, block uniform b
    const int b = flat / N;
    const int n = flat - b * N;

    const float qx = qxyz[(b * 3 + 0) * N + n];
    const float qy = qxyz[(b * 3 + 1) * N + n];
    const float qz = qxyz[(b * 3 + 2) * N + n];

    float bd[KNN_K];
    int   bi[KNN_K];
#pragma unroll
    for (int i = 0; i < KNN_K; ++i) { bd[i] = 3.402823466e+38f; bi[i] = 0; }

    __shared__ float sx[128], sy[128], sz[128];

    for (int base = 0; base < N; base += 128) {
        if (threadIdx.x < 128) {
            int j = base + threadIdx.x;
            sx[threadIdx.x] = rxyz[(b * 3 + 0) * N + j];
            sy[threadIdx.x] = rxyz[(b * 3 + 1) * N + j];
            sz[threadIdx.x] = rxyz[(b * 3 + 2) * N + j];
        }
        __syncthreads();
        for (int jj = 0; jj < 128; ++jj) {
            float dx = sx[jj] - qx;
            float dy = sy[jj] - qy;
            float dz = sz[jj] - qz;
            float d  = dx * dx + dy * dy + dz * dz;
            if (d < bd[KNN_K - 1]) {
                float cd = d; int ci = base + jj;
#pragma unroll
                for (int s = 0; s < KNN_K; ++s) {
                    if (cd < bd[s]) {
                        float td = bd[s]; int ti = bi[s];
                        bd[s] = cd; bi[s] = ci;
                        cd = td; ci = ti;
                    }
                }
            }
        }
        __syncthreads();
    }
#pragma unroll
    for (int i = 0; i < KNN_K; ++i) outidx[(size_t)flat * KNN_K + i] = bi[i];
}

// ---------------------------------------------------------------------------
// A-fragment (16x32 f16) load from row-major LDS: two contiguous 16B runs.
// lanes 0-15 row M=lane hold K {kb..kb+15};
// lanes 16-31 same rows hold K {kb+16..kb+31}
// ---------------------------------------------------------------------------
__device__ inline v16h load_a16v(const _Float16* __restrict__ X, int lane, int kb)
{
    const int row = lane & 15;
    const int hs  = lane >> 4;
    const v8h* p = (const v8h*)(X + row * DIM_PAD + kb + hs * 16);
    v8h lo = p[0];
    v8h hi = p[1];
    return __builtin_shufflevector(lo, hi, 0, 1, 2, 3, 4, 5, 6, 7,
                                           8, 9, 10, 11, 12, 13, 14, 15);
}

// ---------------------------------------------------------------------------
// Kernel 2: grouped feature build + 2-layer MLP via WMMA.
// One wave handles one point n (its 16 neighbors = 16 M-rows).
// Weights staged in LDS pre-swizzled to B-fragment register order so each
// lane's fragment is one aligned v16h load. Block = 128 (4 waves).
// feat out layout [B,N,K,Cm] f32.
// ---------------------------------------------------------------------------
__global__ void __launch_bounds__(128)
mlp_feat_kernel(const float* __restrict__ points1, const float* __restrict__ points2,
                const float* __restrict__ xyz1,   const float* __restrict__ xyzr,
                const int*   __restrict__ idx,
                const float* __restrict__ W0, const float* __restrict__ b0,
                const float* __restrict__ W1, const float* __restrict__ b1,
                float* __restrict__ feat, int N)
{
    __shared__ v16h sW0f[5 * 4 * 32];            // [kc][ntile][lane] fragments
    __shared__ v16h sW1f[2 * 4 * 32];
    __shared__ alignas(16) _Float16 sX[4][16 * DIM_PAD];  // per-wave A staging

    const int tid  = threadIdx.x;
    const int wave = tid >> 5;
    const int lane = tid & 31;

    // Stage W0^T fragments: lane holds 16 consecutive c at fixed o
    for (int task = tid; task < 5 * 4 * 32; task += 128) {
        const int ln   = task & 31;
        const int frag = task >> 5;          // kc*4 + t
        const int kc = frag >> 2, t = frag & 3;
        const int o  = t * 16 + (ln & 15);
        const int ks = kc * 32 + (ln >> 4) * 16;
        v16h w;
#pragma unroll
        for (int i = 0; i < 16; ++i) {
            int cc = ks + i;
            w[i] = (cc < DIM_IN) ? (_Float16)W0[o * DIM_IN + cc] : (_Float16)0.0f;
        }
        sW0f[task] = w;
    }
    for (int task = tid; task < 2 * 4 * 32; task += 128) {
        const int ln   = task & 31;
        const int frag = task >> 5;
        const int kc = frag >> 2, t = frag & 3;
        const int o  = t * 16 + (ln & 15);
        const int ks = kc * 32 + (ln >> 4) * 16;
        v16h w;
#pragma unroll
        for (int i = 0; i < 16; ++i) w[i] = (_Float16)W1[o * CM + ks + i];
        sW1f[task] = w;
    }

    const int flat = blockIdx.x * 4 + wave;     // b*N + n
    const int b = flat / N;
    const int n = flat - b * N;
    const int* idxp = idx + (size_t)flat * KNN_K;
    _Float16* X = sX[wave];

    // Stage A rows: X[row=k][c] = [ f1[n] | f2[idx[k]] | dir_xyz | 0-pad ]
    // 8 halfs per vector store; runs never cross rows (160 % 8 == 0)
    for (int t0 = lane * 8; t0 < 16 * DIM_PAD; t0 += 32 * 8) {
        const int row = t0 / DIM_PAD;
        const int c0  = t0 - row * DIM_PAD;
        const int j   = idxp[row];
        v8h out;
#pragma unroll
        for (int i = 0; i < 8; ++i) {
            const int c = c0 + i;
            float v = 0.0f;
            if (c < 64) {
                v = points1[(b * 64 + c) * N + n];
            } else if (c < 128) {
                v = points2[(b * 64 + (c - 64)) * N + j];
            } else if (c < DIM_IN) {
                const int cc = c - 128;
                v = xyzr[(b * 3 + cc) * N + j] - xyz1[(b * 3 + cc) * N + n];
            }
            out[i] = (_Float16)v;
        }
        *(v8h*)(X + t0) = out;
    }
    __syncthreads();

    // ---- Layer 0: [16 x 160] x [160 x 64] -> [16 x 64], leaky relu ----
    v8f acc0[4] = {};
#pragma unroll
    for (int kc = 0; kc < DIM_PAD / 32; ++kc) {
        v16h a = load_a16v(X, lane, kc * 32);
#pragma unroll
        for (int t = 0; t < 4; ++t) {
            v16h bf = sW0f[(kc * 4 + t) * 32 + lane];
            acc0[t] = __builtin_amdgcn_wmma_f32_16x16x32_f16(
                false, a, false, bf, (short)0, acc0[t], false, false);
        }
    }

    const int colo  = lane & 15;
    const int rbase = (lane >> 4) * 8;
    // bias + leaky, write H (f16) back into this wave's X region
#pragma unroll
    for (int t = 0; t < 4; ++t) {
        const int o = t * 16 + colo;
        const float bo = b0[o];
#pragma unroll
        for (int v = 0; v < 8; ++v) {
            float x = acc0[t][v] + bo;
            x = (x >= 0.0f) ? x : LEAKY_S * x;
            X[(rbase + v) * DIM_PAD + o] = (_Float16)x;
        }
    }
    __syncthreads();   // cross-lane LDS ordering before layer-1 A loads

    // ---- Layer 1: [16 x 64] x [64 x 64] -> [16 x 64], leaky relu ----
    v8f acc1[4] = {};
#pragma unroll
    for (int kc = 0; kc < 2; ++kc) {
        v16h a = load_a16v(X, lane, kc * 32);
#pragma unroll
        for (int t = 0; t < 4; ++t) {
            v16h bf = sW1f[(kc * 4 + t) * 32 + lane];
            acc1[t] = __builtin_amdgcn_wmma_f32_16x16x32_f16(
                false, a, false, bf, (short)0, acc1[t], false, false);
        }
    }

#pragma unroll
    for (int t = 0; t < 4; ++t) {
        const int o = t * 16 + colo;
        const float bo = b1[o];
#pragma unroll
        for (int v = 0; v < 8; ++v) {
            const int k = rbase + v;
            float x = acc1[t][v] + bo;
            x = (x >= 0.0f) ? x : LEAKY_S * x;
            feat[((size_t)flat * KNN_K + k) * CM + o] = x;
        }
    }
}

// ---------------------------------------------------------------------------
// Kernel 3 (WMMA): wqk[b,c,k,l] = sum_n feat[b,n,k,c] * featw[b,n,l,c]
// One wave per (b,c): 16x16 GEMM with K = N, chunked by 32, f32 accumulate.
// A chunk = feat (rows k, K = n); B chunk = featw (K = n, cols l).
// ---------------------------------------------------------------------------
__global__ void __launch_bounds__(128)
attn_wqk_wmma_kernel(const float* __restrict__ feat, const float* __restrict__ featw,
                     float* __restrict__ wqk, int N)
{
    const int wid  = (blockIdx.x * 128 + threadIdx.x) >> 5;   // b*64 + c
    const int lane = threadIdx.x & 31;
    const int b = wid >> 6;
    const int c = wid & 63;
    const int row = lane & 15;   // A: k-row ; B: l-column
    const int hs  = lane >> 4;

    v8f acc = {};
    for (int n0 = 0; n0 < N; n0 += 32) {
        v16h a, bf;
#pragma unroll
        for (int i = 0; i < 8; ++i) {
            const int nn0 = n0 + hs * 8 + i;
            const int nn1 = n0 + 16 + hs * 8 + i;
            a[i]     = (_Float16)feat[(((size_t)b * N + nn0) * KNN_K + row) * CM + c];
            a[8 + i] = (_Float16)feat[(((size_t)b * N + nn1) * KNN_K + row) * CM + c];
        }
#pragma unroll
        for (int i = 0; i < 16; ++i) {
            const int nn = n0 + (i & 8) * 2 + hs * 8 + (i & 7);
            bf[i] = (_Float16)featw[(((size_t)b * N + nn) * KNN_K + row) * CM + c];
        }
        acc = __builtin_amdgcn_wmma_f32_16x16x32_f16(
            false, a, false, bf, (short)0, acc, false, false);
    }
    // D: lane col = l = lane&15, VGPR v -> row k = v + hs*8
#pragma unroll
    for (int v = 0; v < 8; ++v)
        wqk[((size_t)wid * 16 + (v + hs * 8)) * 16 + (lane & 15)] = acc[v];
}

// ---------------------------------------------------------------------------
// Kernel 4: softmax over l, /sqrt(3), clip(1e-10), then
//   S1[b,c,k] = sum_l p   (wqk.sum(-1))
//   S2[b,c,l] = sum_k p   (wqk.sum(-2))
// ---------------------------------------------------------------------------
__global__ void __launch_bounds__(256)
softmax_sums_kernel(const float* __restrict__ wqk,
                    float* __restrict__ S1, float* __restrict__ S2)
{
    const int bc = blockIdx.x;
    const int tid = threadIdx.x;
    const int k = tid >> 4;
    const int l = tid & 15;

    __shared__ float p[16][16];

    float v = wqk[(size_t)bc * 256 + tid];
    p[k][l] = v;
    __syncthreads();
    float m = p[k][0];
#pragma unroll
    for (int i = 1; i < 16; ++i) m = fmaxf(m, p[k][i]);
    __syncthreads();
    float e = __expf(v - m);
    p[k][l] = e;
    __syncthreads();
    float s = 0.0f;
#pragma unroll
    for (int i = 0; i < 16; ++i) s += p[k][i];
    float prob = fmaxf(e / s * INV_SQRT3, 1e-10f);
    __syncthreads();
    p[k][l] = prob;
    __syncthreads();
    if (l == 0) {
        float t = 0.0f;
#pragma unroll
        for (int i = 0; i < 16; ++i) t += p[k][i];
        S1[bc * 16 + k] = t;
    }
    if (k == 0) {
        float t = 0.0f;
#pragma unroll
        for (int i = 0; i < 16; ++i) t += p[i][l];
        S2[bc * 16 + l] = t;
    }
}

// ---------------------------------------------------------------------------
// Kernel 5: combined[b,n,c] = sum_l feat[b,n,l,c]*S2[b,c,l]
//                           + sum_l featw[b,n,l,c]*S1[b,c,l]
// ---------------------------------------------------------------------------
__global__ void __launch_bounds__(256)
cost_combined_kernel(const float* __restrict__ feat, const float* __restrict__ featw,
                     const float* __restrict__ S1,   const float* __restrict__ S2,
                     float* __restrict__ combined, int N)
{
    const int tid  = threadIdx.x;
    const int flat = blockIdx.x * 4 + (tid >> 6);   // b*N + n
    const int c    = tid & 63;
    const int b    = flat / N;

    const float* S1p = S1 + (b * 64 + c) * 16;
    const float* S2p = S2 + (b * 64 + c) * 16;

    float acc = 0.0f;
#pragma unroll
    for (int l = 0; l < 16; ++l) {
        acc += feat [((size_t)flat * KNN_K + l) * CM + c] * S2p[l];
        acc += featw[((size_t)flat * KNN_K + l) * CM + c] * S1p[l];
    }
    combined[(size_t)flat * CM + c] = acc;
}

// ---------------------------------------------------------------------------
// Kernel 6: WeightNet (3->8->8->64, BN eval + relu) + gather/weighted sum.
// grid = B*N blocks of 64 threads. out layout [B,Cm,N]
// ---------------------------------------------------------------------------
__global__ void __launch_bounds__(64)
final_kernel(const float* __restrict__ xyz1, const int* __restrict__ idxs,
             const float* __restrict__ combined,
             const float* __restrict__ wW0, const float* __restrict__ wb0,
             const float* __restrict__ wg0, const float* __restrict__ wbe0,
             const float* __restrict__ wW1, const float* __restrict__ wb1,
             const float* __restrict__ wg1, const float* __restrict__ wbe1,
             const float* __restrict__ wW2, const float* __restrict__ wb2,
             const float* __restrict__ wg2, const float* __restrict__ wbe2,
             float* __restrict__ out, int N)
{
    const int flat = blockIdx.x;           // b*N + n
    const int b = flat / N;
    const int n = flat - b * N;
    const int tid = threadIdx.x;

    __shared__ float h1s[16][8];
    __shared__ int   js[16];

    if (tid < 16) {
        const int k = tid;
        const int j = idxs[(size_t)flat * KNN_K + k];
        js[k] = j;
        float d0 = xyz1[(b * 3 + 0) * N + j] - xyz1[(b * 3 + 0) * N + n];
        float d1 = xyz1[(b * 3 + 1) * N + j] - xyz1[(b * 3 + 1) * N + n];
        float d2 = xyz1[(b * 3 + 2) * N + j] - xyz1[(b * 3 + 2) * N + n];
        float h0[8];
#pragma unroll
        for (int o = 0; o < 8; ++o) {
            float x = wW0[o * 3 + 0] * d0 + wW0[o * 3 + 1] * d1 + wW0[o * 3 + 2] * d2 + wb0[o];
            x = x * (wg0[o] * BN_SCALE) + wbe0[o];
            h0[o] = fmaxf(x, 0.0f);
        }
#pragma unroll
        for (int o = 0; o < 8; ++o) {
            float x = wb1[o];
#pragma unroll
            for (int i = 0; i < 8; ++i) x += wW1[o * 8 + i] * h0[i];
            x = x * (wg1[o] * BN_SCALE) + wbe1[o];
            h1s[k][o] = fmaxf(x, 0.0f);
        }
    }
    __syncthreads();

    const int c = tid;
    const float sc2   = wg2[c] * BN_SCALE;
    const float sh2   = wbe2[c];
    const float bias2 = wb2[c];
    float w2r[8];
#pragma unroll
    for (int i = 0; i < 8; ++i) w2r[i] = wW2[c * 8 + i];

    float acc = 0.0f;
#pragma unroll
    for (int k = 0; k < 16; ++k) {
        float x = bias2;
#pragma unroll
        for (int i = 0; i < 8; ++i) x += w2r[i] * h1s[k][i];
        float wv = fmaxf(x * sc2 + sh2, 0.0f);
        acc += wv * combined[((size_t)b * N + js[k]) * CM + c];
    }
    out[((size_t)b * CM + c) * N + n] = acc;
}

// ---------------------------------------------------------------------------
// Launcher
// ---------------------------------------------------------------------------
extern "C" void kernel_launch(void* const* d_in, const int* in_sizes, int n_in,
                              void* d_out, int out_size, void* d_ws, size_t ws_size,
                              hipStream_t stream) {
    const float* xyz1    = (const float*)d_in[0];
    const float* xyz2    = (const float*)d_in[1];
    const float* xyz2w   = (const float*)d_in[2];
    const float* points1 = (const float*)d_in[3];
    const float* points2 = (const float*)d_in[4];
    const float* mlp_W0  = (const float*)d_in[5];
    const float* mlp_b0  = (const float*)d_in[6];
    const float* mlp_W1  = (const float*)d_in[7];
    const float* mlp_b1  = (const float*)d_in[8];
    const float* wn_W0   = (const float*)d_in[9];
    const float* wn_b0   = (const float*)d_in[10];
    const float* wn_g0   = (const float*)d_in[11];
    const float* wn_be0  = (const float*)d_in[12];
    const float* wn_W1   = (const float*)d_in[13];
    const float* wn_b1   = (const float*)d_in[14];
    const float* wn_g1   = (const float*)d_in[15];
    const float* wn_be1  = (const float*)d_in[16];
    const float* wn_W2   = (const float*)d_in[17];
    const float* wn_b2   = (const float*)d_in[18];
    const float* wn_g2   = (const float*)d_in[19];
    const float* wn_be2  = (const float*)d_in[20];

    const int B = 2;
    const int N = in_sizes[0] / (3 * B);   // 4096

    // Workspace carve-up
    char* ws = (char*)d_ws;
    size_t off = 0;
    auto carve = [&](size_t bytes) -> void* {
        void* p = ws + off;
        off = (off + bytes + 255) & ~(size_t)255;
        return p;
    };
    const size_t idx_bytes  = (size_t)B * N * KNN_K * sizeof(int);
    const size_t feat_bytes = (size_t)B * N * KNN_K * CM * sizeof(float);

    int*   idx2     = (int*)  carve(idx_bytes);
    int*   idxw     = (int*)  carve(idx_bytes);
    int*   idxs     = (int*)  carve(idx_bytes);
    float* feat     = (float*)carve(feat_bytes);
    float* featw    = (float*)carve(feat_bytes);
    float* wqk      = (float*)carve((size_t)B * CM * 256 * sizeof(float));
    float* S1       = (float*)carve((size_t)B * CM * 16 * sizeof(float));
    float* S2       = (float*)carve((size_t)B * CM * 16 * sizeof(float));
    float* combined = (float*)carve((size_t)B * N * CM * sizeof(float));

    const int knn_blocks = (B * N) / 256;
    knn_kernel<<<knn_blocks, 256, 0, stream>>>(xyz1, xyz2,  idx2, N);
    knn_kernel<<<knn_blocks, 256, 0, stream>>>(xyz1, xyz2w, idxw, N);
    knn_kernel<<<knn_blocks, 256, 0, stream>>>(xyz1, xyz1,  idxs, N);

    const int mlp_blocks = (B * N) / 4;
    mlp_feat_kernel<<<mlp_blocks, 128, 0, stream>>>(
        points1, points2, xyz1, xyz2,  idx2, mlp_W0, mlp_b0, mlp_W1, mlp_b1, feat,  N);
    mlp_feat_kernel<<<mlp_blocks, 128, 0, stream>>>(
        points1, points2, xyz1, xyz2w, idxw, mlp_W0, mlp_b0, mlp_W1, mlp_b1, featw, N);

    attn_wqk_wmma_kernel<<<(B * CM) / 4, 128, 0, stream>>>(feat, featw, wqk, N);
    softmax_sums_kernel<<<B * CM, 256, 0, stream>>>(wqk, S1, S2);
    cost_combined_kernel<<<(B * N) / 4, 256, 0, stream>>>(feat, featw, S1, S2, combined, N);

    final_kernel<<<B * N, 64, 0, stream>>>(
        xyz1, idxs, combined,
        wn_W0, wn_b0, wn_g0, wn_be0,
        wn_W1, wn_b1, wn_g1, wn_be1,
        wn_W2, wn_b2, wn_g2, wn_be2,
        (float*)d_out, N);
}
